// BERT_TUCKER_82867099009271
// MI455X (gfx1250) — matrix operand order repaired and checked by reading, not compiled, not run on hardware
//
#include <hip/hip_runtime.h>
#include <hip/hip_bf16.h>
#include <stdint.h>

// Problem constants (match reference)
#define B_   16
#define S_   512
#define H_   768
#define TS_  20
#define IS_  20
#define D_   808            // H + TS + IS
#define M_   36
#define E_   12
#define RN_  97
#define D2_  50
#define NW_  40400          // D2 * D  (columns of the big GEMM)
#define KPAD 832            // 808 padded to multiple of 32
#define NPAD 40448          // 40400 padded to multiple of 128/64
#define MROWS 192           // B * E distinct head rows
#define MT_WG 96            // M tile per workgroup (half of 192)
#define NT_WG 128           // N tile per workgroup (8 waves x 16)
#define KLDS 840            // A row stride in LDS (bf16 elems): 840*2B/4 % 64 = 36 -> conflict-free frags

typedef __attribute__((ext_vector_type(16))) __bf16 v16bf;
typedef __attribute__((ext_vector_type(8)))  float  v8f;
typedef int v4i_ __attribute__((vector_size(4 * sizeof(int))));
typedef __attribute__((address_space(1))) v4i_* gas_i4p;   // global int4*
typedef __attribute__((address_space(3))) v4i_* las_i4p;   // LDS int4*

union Frag { uint4 q[2]; v16bf v; };

#define HAS_ASYNC_LDS __has_builtin(__builtin_amdgcn_global_load_async_to_lds_b128)

// ---------------------------------------------------------------------------
// K0: W fp32 [K=808, N=40400] (flat a*40400+n) -> Wt bf16 [NPAD, KPAD]
//     row-major (K contiguous per row), zero padded in both dims.
// ---------------------------------------------------------------------------
__global__ void k0_convert_W(const float* __restrict__ Wf,
                             __hip_bfloat16* __restrict__ Wt) {
  __shared__ float tile[32][33];
  int n0 = blockIdx.x * 32;
  int k0 = blockIdx.y * 32;
  int tx = threadIdx.x;   // 0..31
  int ty = threadIdx.y;   // 0..7
#pragma unroll
  for (int i = 0; i < 4; ++i) {
    int k = k0 + ty + i * 8;
    int n = n0 + tx;
    float v = 0.0f;
    if (k < D_ && n < NW_) v = Wf[(size_t)k * NW_ + n];
    tile[ty + i * 8][tx] = v;
  }
  __syncthreads();
#pragma unroll
  for (int i = 0; i < 4; ++i) {
    int n = n0 + ty + i * 8;   // < NPAD by grid construction
    int k = k0 + tx;           // < KPAD by grid construction
    Wt[(size_t)n * KPAD + k] = __float2bfloat16(tile[tx][ty + i * 8]);
  }
}

// ---------------------------------------------------------------------------
// K1: mention pooling. x[B, M+1, D]; node 0 = cls (hidden[:,0,:] ++ zeros).
// ---------------------------------------------------------------------------
__global__ void k1_mention_pool(const float* __restrict__ hidden,
                                const int* __restrict__ etype,
                                const int* __restrict__ eid,
                                const int* __restrict__ mid,
                                const float* __restrict__ type_emb,
                                const float* __restrict__ id_emb,
                                float* __restrict__ x) {
  int b = blockIdx.x;
  int chunk = blockIdx.y;         // 0..3
  int tid = threadIdx.x;          // 0..255
  int dd = chunk * 256 + tid;
  bool valid = dd < D_;

  __shared__ int   midL[S_];
  __shared__ int   etL[S_];
  __shared__ int   eiL[S_];
  __shared__ float acc[M_][256];
  __shared__ float cinv[M_];

  for (int s = tid; s < S_; s += 256) {
    midL[s] = mid[b * S_ + s];
    etL[s]  = etype[b * S_ + s];
    eiL[s]  = eid[b * S_ + s];
  }
  for (int m = 0; m < M_; ++m) acc[m][tid] = 0.0f;
  __syncthreads();

  if (valid) {
    for (int s = 0; s < S_; ++s) {
      int m = midL[s];
      if (m > 0) {
        float v;
        if (dd < H_)             v = hidden[((size_t)b * S_ + s) * H_ + dd];
        else if (dd < H_ + TS_)  v = type_emb[etL[s] * TS_ + (dd - H_)];
        else                     v = id_emb[eiL[s] * IS_ + (dd - H_ - TS_)];
        acc[m - 1][tid] += v;
      }
    }
  }
  if (tid < M_) {
    int c = 0;
    for (int s = 0; s < S_; ++s) c += (midL[s] == tid + 1);
    cinv[tid] = (c > 0) ? 1.0f / (float)c : 0.0f;
  }
  __syncthreads();

  if (valid) {
    float cls = (dd < H_) ? hidden[(size_t)b * S_ * H_ + dd] : 0.0f;
    x[((size_t)b * (M_ + 1) + 0) * D_ + dd] = cls;
    for (int m = 0; m < M_; ++m)
      x[((size_t)b * (M_ + 1) + m + 1) * D_ + dd] = acc[m][tid] * cinv[m];
  }
}

// ---------------------------------------------------------------------------
// K2: entity pooling ent[B,E,D] = normalize(tbl[b,e+1,:]) @ x[b,:,:]
//     Also emit bf16 A-matrix entb [192, KPAD] (zero padded K).
// ---------------------------------------------------------------------------
__global__ void k2_entity_pool(const float* __restrict__ x,
                               const float* __restrict__ tbl,
                               float* __restrict__ ent,
                               __hip_bfloat16* __restrict__ entb) {
  int b = blockIdx.x;
  int e = blockIdx.y;             // 0..11  (row e+1 of table; node 0 dropped)
  int tid = threadIdx.x;          // 0..255
  const float* row = tbl + ((size_t)b * (E_ + 1) + (e + 1)) * (M_ + 1);
  float wsum = 0.0f;
  for (int m = 0; m <= M_; ++m) wsum += row[m];
  float winv = (wsum > 0.0f) ? 1.0f / wsum : 0.0f;

  for (int dd = tid; dd < KPAD; dd += 256) {
    float v = 0.0f;
    if (dd < D_) {
      float s = 0.0f;
      for (int m = 0; m <= M_; ++m)
        s += row[m] * x[((size_t)b * (M_ + 1) + m) * D_ + dd];
      v = s * winv;
      ent[((size_t)b * E_ + e) * D_ + dd] = v;
    }
    entb[((size_t)b * E_ + e) * KPAD + dd] = __float2bfloat16(v);
  }
}

// ---------------------------------------------------------------------------
// K3: G[192, 40400] = A[192,808] * Wv[808,40400]   (bf16 WMMA, fp32 acc)
//   WG = [96(M) x 128(N)]; 8 waves, wave w owns N sub-tile 16w..16w+15.
//   A half staged in LDS (async-to-LDS), fragments via ds_load_b128 (DScnt).
//   B streamed from global, register double-buffered, prefetch distance 2
//   chunks so WMMAs overlap the HBM stream (LOADcnt waits stay > 0).
// ---------------------------------------------------------------------------
__device__ __forceinline__ void load_bfrag(Frag& f, const __hip_bfloat16* p) {
  f.q[0] = *reinterpret_cast<const uint4*>(p);       // K base..base+7   -> VGPR 0-3
  f.q[1] = *reinterpret_cast<const uint4*>(p + 8);   // K base+8..base+15 -> VGPR 4-7
}

__global__ void __launch_bounds__(256)
k3_gemm(const __hip_bfloat16* __restrict__ A,     // entb [192, KPAD]
        const __hip_bfloat16* __restrict__ Bt,    // Wt   [NPAD, KPAD] = B^T
        float* __restrict__ G) {                  // [192, NW_]
  extern __shared__ __align__(16) char smem_raw[];
  __hip_bfloat16* As = reinterpret_cast<__hip_bfloat16*>(smem_raw);  // [96][KLDS]

  int tid  = threadIdx.x;
  int lane = tid & 31;
  int w    = tid >> 5;                     // 0..7
  int nsup = blockIdx.x >> 1;              // N super-tile
  int M0   = (blockIdx.x & 1) * MT_WG;     // 0 or 96 (pairs share N tile -> L2 reuse of W)
  int N0   = nsup * NT_WG + w * 16;
  int laneHi = lane >> 4;                  // 0/1
  int lane16 = lane & 15;

  // ---- stage A half [96 x 832] into LDS (row stride KLDS) ----
  for (int idx = tid; idx < MT_WG * (KPAD / 16); idx += 256) {
    int r = idx / (KPAD / 16);
    int c = (idx % (KPAD / 16)) * 16;
    const __hip_bfloat16* src = A + (size_t)(M0 + r) * KPAD + c;
    __hip_bfloat16* dst = As + r * KLDS + c;
#if HAS_ASYNC_LDS
    __builtin_amdgcn_global_load_async_to_lds_b128((gas_i4p)src, (las_i4p)dst, 0, 0);
#else
    *reinterpret_cast<uint4*>(dst) = *reinterpret_cast<const uint4*>(src);
#endif
  }
#if HAS_ASYNC_LDS
#if __has_builtin(__builtin_amdgcn_s_wait_asynccnt)
  __builtin_amdgcn_s_wait_asynccnt(0);
#else
  asm volatile("s_wait_asynccnt 0x0" ::: "memory");
#endif
#endif
  __syncthreads();

  v8f acc[6];
#pragma unroll
  for (int i = 0; i < 6; ++i) acc[i] = (v8f){0, 0, 0, 0, 0, 0, 0, 0};

  // B fragment base: column n = N0+lane16; lanes>=16 take K offset +16.
  const __hip_bfloat16* bbase = Bt + (size_t)(N0 + lane16) * KPAD + laneHi * 16;
  const int aoff = laneHi * 16;            // A frag: chunks at k0+c and k0+c+8

#define WMMA_SET(BF, K0)                                                      \
  {                                                                           \
    _Pragma("unroll")                                                         \
    for (int mt = 0; mt < 6; ++mt) {                                          \
      const __hip_bfloat16* ap = As + (mt * 16 + lane16) * KLDS + (K0) + aoff;\
      Frag af;                                                                \
      af.q[0] = *reinterpret_cast<const uint4*>(ap);                          \
      af.q[1] = *reinterpret_cast<const uint4*>(ap + 8);                      \
      acc[mt] = __builtin_amdgcn_wmma_f32_16x16x32_bf16(                      \
          false, af.v, false, BF.v, (short)0, acc[mt], false, false);         \
    }                                                                         \
  }

  Frag b0, b1;
  load_bfrag(b0, bbase + 0);     // chunk 0 in flight
  load_bfrag(b1, bbase + 32);    // chunk 1 in flight  (prefetch depth 2)

  for (int k0 = 0; k0 < KPAD; k0 += 64) {
    WMMA_SET(b0, k0);                                  // waits b0; b1 + newer stay outstanding
    int kn0 = k0 + 64;  if (kn0 >= KPAD) kn0 = 0;      // tail: harmless dummy refill
    load_bfrag(b0, bbase + kn0);
    WMMA_SET(b1, k0 + 32);                             // waits b1; b0' stays outstanding
    int kn1 = k0 + 96;  if (kn1 >= KPAD) kn1 = 0;
    load_bfrag(b1, bbase + kn1);
  }
#undef WMMA_SET

  int col = N0 + lane16;
  if (col < NW_) {
#pragma unroll
    for (int mt = 0; mt < 6; ++mt) {
#pragma unroll
      for (int r = 0; r < 8; ++r) {
        int mrow = M0 + mt * 16 + r + laneHi * 8;  // C/D layout: VGPR r -> M=r(+8)
        G[(size_t)mrow * NW_ + col] = acc[mt][r];
      }
    }
  }
}

// ---------------------------------------------------------------------------
// K4: z[b,p,r] = G[(b,k), r*808+j'] . ent[b,j]  ->  BN  ->  scores = z @ R^T
// ---------------------------------------------------------------------------
__global__ void __launch_bounds__(256)
k4_score(const float* __restrict__ G,
         const float* __restrict__ ent,
         const float* __restrict__ Rm,
         const float* __restrict__ gamma, const float* __restrict__ beta,
         const float* __restrict__ mean,  const float* __restrict__ var,
         float* __restrict__ out) {
  int b = blockIdx.y;
  int p = blockIdx.x;              // 0..143
  int k = p / E_;
  int j = p % E_;
  int tid = threadIdx.x;
  int lane = tid & 31, w = tid >> 5;

  __shared__ float tail[D_];
  __shared__ float zb[D2_];
  for (int d = tid; d < D_; d += 256)
    tail[d] = ent[((size_t)b * E_ + j) * D_ + d];
  __syncthreads();

  const float* grow = G + (size_t)(b * E_ + k) * NW_;
  for (int rr = w; rr < D2_; rr += 8) {
    float s = 0.0f;
    const float* gseg = grow + rr * D_;
    for (int d = lane; d < D_; d += 32) s += gseg[d] * tail[d];
#pragma unroll
    for (int off = 16; off > 0; off >>= 1) s += __shfl_xor(s, off, 32);
    if (lane == 0) zb[rr] = s;
  }
  __syncthreads();
  if (tid < D2_) {
    float z = zb[tid];
    z = (z - mean[tid]) * rsqrtf(var[tid] + 1e-5f) * gamma[tid] + beta[tid];
    zb[tid] = z;
  }
  __syncthreads();
  if (tid < RN_) {
    float s = 0.0f;
#pragma unroll
    for (int r = 0; r < D2_; ++r) s += zb[r] * Rm[tid * D2_ + r];
    out[((size_t)b * (E_ * E_) + p) * RN_ + tid] = s;
  }
}

// ---------------------------------------------------------------------------
extern "C" void kernel_launch(void* const* d_in, const int* in_sizes, int n_in,
                              void* d_out, int out_size, void* d_ws, size_t ws_size,
                              hipStream_t stream) {
  (void)in_sizes; (void)n_in; (void)out_size; (void)ws_size;
  const float* hidden = (const float*)d_in[0];
  const int*   etype  = (const int*)d_in[1];
  const int*   eid    = (const int*)d_in[2];
  const int*   mid    = (const int*)d_in[3];
  const float* tbl    = (const float*)d_in[4];
  const float* temb   = (const float*)d_in[5];
  const float* iemb   = (const float*)d_in[6];
  const float* Wf     = (const float*)d_in[7];
  const float* Rm     = (const float*)d_in[8];
  const float* g1     = (const float*)d_in[9];
  const float* b1     = (const float*)d_in[10];
  const float* m1     = (const float*)d_in[11];
  const float* v1     = (const float*)d_in[12];

  // Workspace layout (~101 MB total), 256-B aligned slabs.
  char* ws = (char*)d_ws;
  size_t off = 0;
  auto take = [&](size_t bytes) {
    size_t o = off;
    off = (off + bytes + 255) & ~(size_t)255;
    return o;
  };
  __hip_bfloat16* Wt   = (__hip_bfloat16*)(ws + take((size_t)NPAD * KPAD * 2)); // 67.3 MB
  float*          G    = (float*)         (ws + take((size_t)MROWS * NW_ * 4)); // 31.0 MB
  float*          x    = (float*)         (ws + take((size_t)B_ * (M_ + 1) * D_ * 4));
  float*          ent  = (float*)         (ws + take((size_t)B_ * E_ * D_ * 4));
  __hip_bfloat16* entb = (__hip_bfloat16*)(ws + take((size_t)MROWS * KPAD * 2));

  k0_convert_W   <<<dim3(NPAD / 32, KPAD / 32), dim3(32, 8), 0, stream>>>(Wf, Wt);
  k1_mention_pool<<<dim3(B_, 4), 256, 0, stream>>>(hidden, etype, eid, mid, temb, iemb, x);
  k2_entity_pool <<<dim3(B_, E_), 256, 0, stream>>>(x, tbl, ent, entb);
  // 632 WGs: (N super-tile, M half) pairs adjacent so W streams HBM ~once.
  k3_gemm<<<dim3((NPAD / NT_WG) * 2), 256, MT_WG * KLDS * 2, stream>>>(entb, Wt, G);
  k4_score<<<dim3(E_ * E_, B_), 256, 0, stream>>>(G, ent, Rm, g1, b1, m1, v1,
                                                  (float*)d_out);
}
